// Self_Attention_Layer_AfterCat_75024488726904
// MI455X (gfx1250) — matrix operand order repaired and checked by reading, not compiled, not run on hardware
//
#include <hip/hip_runtime.h>

// ---------------------------------------------------------------------------
// Self-attention (post-concat bi-RNN): B=4, S=4096, D=1024.
//   Q = X Wq^T, K = X Wk^T, V = X Wv^T  (fp32 in, staged as bf16)
//   out = softmax(mask(Q K^T)) V        (flash-style, f32 accumulate)
// All matrix math on v_wmma_f32_16x16x32_bf16 (CDNA5, wave32).
// Q-block staging via global_load_async_to_lds_b128 (ASYNCcnt path).
// Workspace: Qb[B*S*D] bf16 + Kb[B*S*D] bf16 + Vt[B*D*S] bf16 = 96 MB.
// ---------------------------------------------------------------------------

#define B_DIM 4
#define S_LEN 4096
#define D_DIM 1024

typedef __bf16 bf16_t;
typedef __attribute__((ext_vector_type(16))) bf16_t v16bf;
typedef __attribute__((ext_vector_type(8)))  float  v8f;

__device__ inline unsigned short f2bf(float f) {  // fp32 -> bf16 RNE
  unsigned int u = __float_as_uint(f);
  u += 0x7FFFu + ((u >> 16) & 1u);
  return (unsigned short)(u >> 16);
}
__device__ inline unsigned int pack2(float a, float b) {
  return (unsigned int)f2bf(a) | ((unsigned int)f2bf(b) << 16);
}

union FragU { uint4 u[2]; v16bf v; };

__device__ inline v16bf frag_ld(const unsigned short* p0, const unsigned short* p1) {
  FragU f;
  f.u[0] = *(const uint4*)p0;   // 8 halfs
  f.u[1] = *(const uint4*)p1;   // 8 halfs
  return f.v;
}
__device__ inline v8f wmma_bf16(v16bf a, v16bf b, v8f c) {
  // (neg_a, A, neg_b, B, c_mod, C, reuse_a, reuse_b)
  return __builtin_amdgcn_wmma_f32_16x16x32_bf16(false, a, false, b, (short)0, c,
                                                 false, false);
}

// Stage 16 consecutive fp32 from global as 16 bf16 into LDS (two b128 stores).
__device__ inline void stage16(const float* __restrict__ g, unsigned short* s) {
  float4 f0 = ((const float4*)g)[0];
  float4 f1 = ((const float4*)g)[1];
  float4 f2 = ((const float4*)g)[2];
  float4 f3 = ((const float4*)g)[3];
  uint4 lo, hi;
  lo.x = pack2(f0.x, f0.y); lo.y = pack2(f0.z, f0.w);
  lo.z = pack2(f1.x, f1.y); lo.w = pack2(f1.z, f1.w);
  hi.x = pack2(f2.x, f2.y); hi.y = pack2(f2.z, f2.w);
  hi.z = pack2(f3.x, f3.y); hi.w = pack2(f3.z, f3.w);
  *(uint4*)s       = lo;
  *(uint4*)(s + 8) = hi;
}

// ---------------------------------------------------------------------------
// Kernel 1: Q/K/V projection GEMM. y[m,n] = sum_d X[m,d] * W[n,d].
// Tile 128x128, BK=32; 8 waves, each wave: 32 rows x 64 cols (2x4 wmma tiles).
// z==2 (V) stores transposed: Vt[b][d][s] (one b128 store per tile: the 8
// accumulator elements of a lane are 8 consecutive s-rows at fixed d).
// ---------------------------------------------------------------------------
#define LDT 40  // LDS tile row stride in halfs (80B, 16B-aligned fragments)

__global__ __launch_bounds__(256) void qkv_proj_kernel(
    const float* __restrict__ X,
    const float* __restrict__ Wq, const float* __restrict__ Wk,
    const float* __restrict__ Wv,
    unsigned short* __restrict__ Qb, unsigned short* __restrict__ Kb,
    unsigned short* __restrict__ Vt)
{
  const int z = blockIdx.z;
  const float* W = (z == 0) ? Wq : ((z == 1) ? Wk : Wv);
  const int m0 = blockIdx.y * 128;
  const int n0 = blockIdx.x * 128;
  const int tid  = threadIdx.x;
  const int wave = tid >> 5, lane = tid & 31;
  const int wm = wave & 3;        // 4 row-groups of 32
  const int wn = wave >> 2;       // 2 col-groups of 64
  const int lr = lane & 15, lh = lane >> 4;

  __shared__ unsigned short Xs[128 * LDT];
  __shared__ unsigned short Ws[128 * LDT];

  v8f acc[2][4];
  for (int mt = 0; mt < 2; ++mt)
    for (int nt = 0; nt < 4; ++nt)
      acc[mt][nt] = v8f{0.f, 0.f, 0.f, 0.f, 0.f, 0.f, 0.f, 0.f};

  const int trow = tid >> 1;           // 0..127
  const int tcol = (tid & 1) * 16;     // 0 or 16

  for (int k0 = 0; k0 < D_DIM; k0 += 32) {
    const float* xg = X + (size_t)(m0 + trow) * D_DIM + k0 + tcol;
    const float* wg = W + (size_t)(n0 + trow) * D_DIM + k0 + tcol;
    if (k0 + 32 < D_DIM) {               // prefetch next k-chunk (L2-resident set)
      __builtin_prefetch(xg + 32, 0, 3);
      __builtin_prefetch(wg + 32, 0, 3);
    }
    stage16(xg, Xs + trow * LDT + tcol);
    stage16(wg, Ws + trow * LDT + tcol);
    __syncthreads();

    v16bf afr[2];
#pragma unroll
    for (int mt = 0; mt < 2; ++mt) {
      const unsigned short* p = Xs + (wm * 32 + mt * 16 + lr) * LDT + lh * 8;
      afr[mt] = frag_ld(p, p + 16);
    }
#pragma unroll
    for (int nt = 0; nt < 4; ++nt) {
      const unsigned short* p = Ws + (wn * 64 + nt * 16 + lr) * LDT + lh * 8;
      v16bf bfr = frag_ld(p, p + 16);
      acc[0][nt] = wmma_bf16(afr[0], bfr, acc[0][nt]);
      acc[1][nt] = wmma_bf16(afr[1], bfr, acc[1][nt]);
    }
    __syncthreads();
  }

  // Epilogue. C layout: elem v of lane L is row M = v + 8*(L>=16), col N = L%16.
  if (z == 2) {
    // V transposed: Vt[b][d][s]; elements v=0..7 are consecutive s.
#pragma unroll
    for (int mt = 0; mt < 2; ++mt) {
#pragma unroll
      for (int nt = 0; nt < 4; ++nt) {
        const int rbase = m0 + wm * 32 + mt * 16 + lh * 8;   // global m = b*S+s
        const int col   = n0 + wn * 64 + nt * 16 + lr;       // d
        const int bb = rbase >> 12, s0 = rbase & (S_LEN - 1);
        uint4 pk;
        pk.x = pack2(acc[mt][nt][0], acc[mt][nt][1]);
        pk.y = pack2(acc[mt][nt][2], acc[mt][nt][3]);
        pk.z = pack2(acc[mt][nt][4], acc[mt][nt][5]);
        pk.w = pack2(acc[mt][nt][6], acc[mt][nt][7]);
        *(uint4*)(Vt + ((size_t)bb << 22) + ((size_t)col << 12) + s0) = pk;
      }
    }
  } else {
    unsigned short* __restrict__ Y = (z == 0) ? Qb : Kb;
#pragma unroll
    for (int mt = 0; mt < 2; ++mt) {
#pragma unroll
      for (int nt = 0; nt < 4; ++nt) {
        const int rbase = m0 + wm * 32 + mt * 16 + lh * 8;
        const int col   = n0 + wn * 64 + nt * 16 + lr;
#pragma unroll
        for (int v = 0; v < 8; ++v)
          Y[(size_t)(rbase + v) * D_DIM + col] = f2bf(acc[mt][nt][v]);
      }
    }
  }
}

// ---------------------------------------------------------------------------
// Kernel 2: flash attention. One workgroup per 16 query rows; 8 waves.
// Key blocks of 128: wave w computes S-subtile for keys [16w,16w+16);
// online softmax (stats via LDS + shfl); wave w owns output d-slice
// [128w, 128w+128) with 8 f32 accum tiles, PV B-fragments streamed from Vt.
// ---------------------------------------------------------------------------
__global__ __launch_bounds__(256) void flash_attn_kernel(
    const unsigned short* __restrict__ Qb, const unsigned short* __restrict__ Kb,
    const unsigned short* __restrict__ Vt, const int* __restrict__ length,
    float* __restrict__ out)
{
  const int b  = blockIdx.y;
  const int q0 = blockIdx.x * 16;
  const int tid  = threadIdx.x;
  const int wave = tid >> 5, lane = tid & 31;
  const int lr = lane & 15, lh = lane >> 4;

  __shared__ unsigned short Qs[16 * D_DIM];   // 32 KB, bf16 query block
  __shared__ float          Ss[16 * 132];     // 8.25 KB, f32 logits (padded)
  __shared__ unsigned short Ps[16 * 128];     // 4 KB, bf16 probabilities

  // ---- async global->LDS copy of the 16x1024 bf16 query block ----
  // GVS mode: 64-bit SGPR base + 32-bit VGPR byte offset; VDST holds the LDS
  // byte address (low 32 bits of the generic shared pointer). ASYNCcnt path.
  {
    const unsigned short* Qg = Qb + ((size_t)b * S_LEN + q0) * D_DIM;
#pragma unroll
    for (int i = 0; i < 8; ++i) {
      const int e = tid * 8 + i * 256 * 8;          // halfs
      const unsigned lds_addr = (unsigned)(uintptr_t)(const void*)(Qs + e);
      asm volatile(
          "global_load_async_to_lds_b128 %0, %1, %2 offset:0"
          :
          : "v"(lds_addr), "v"((unsigned)(e * 2)), "s"(Qg)
          : "memory");
    }
    asm volatile("s_wait_asynccnt 0x0" ::: "memory");
  }
  __syncthreads();

  const int kmax = 2 * length[b];             // valid keys; blocks past are all-masked

  float m_r = -3.0e38f, l_r = 0.0f;           // row stats live in lanes 0..15 (row = lane)
  v8f o[8];
#pragma unroll
  for (int j = 0; j < 8; ++j) o[j] = v8f{0.f, 0.f, 0.f, 0.f, 0.f, 0.f, 0.f, 0.f};

  for (int k0 = 0; k0 < kmax; k0 += 128) {
    const int kq0 = k0 + wave * 16;

    // ---- S subtile = Q(16xD) . K(kq0..kq0+16, D)^T ----
    v8f c = v8f{0.f, 0.f, 0.f, 0.f, 0.f, 0.f, 0.f, 0.f};
    const unsigned short* kg =
        Kb + ((size_t)b * S_LEN + kq0 + lr) * D_DIM + lh * 16;
    if (k0 + 128 < kmax) {
      // prefetch next key block's K row + next Vt line for this lane
      __builtin_prefetch(kg + (size_t)128 * D_DIM, 0, 3);
      __builtin_prefetch(Vt + ((size_t)b << 22) +
                             ((size_t)(wave * 128 + lr * 8) << 12) + k0 + 128,
                         0, 3);
    }
    for (int kc = 0; kc < D_DIM; kc += 32) {
      const unsigned short* qp = Qs + lr * D_DIM + kc + lh * 16;
      v16bf a  = frag_ld(qp, qp + 8);
      v16bf bf = frag_ld(kg + kc, kg + kc + 8);
      c = wmma_bf16(a, bf, c);
    }

    // ---- key-padding mask (column = kq0 + lr for every v) ----
    if (kq0 + lr >= kmax) {
#pragma unroll
      for (int v = 0; v < 8; ++v) c[v] = -3.0e38f;
    }

    // ---- publish logits ----
#pragma unroll
    for (int v = 0; v < 8; ++v)
      Ss[(v + lh * 8) * 132 + wave * 16 + lr] = c[v];
    __syncthreads();

    // ---- online softmax row stats (lanes 0..15, row = lane) ----
    float mnew = m_r, corr = 1.0f;
    if (lane < 16) {
      const float4* sr = (const float4*)(Ss + lane * 132);
      float rmax = -3.0e38f;
      for (int i = 0; i < 32; ++i) {
        float4 t = sr[i];
        rmax = fmaxf(rmax, fmaxf(fmaxf(t.x, t.y), fmaxf(t.z, t.w)));
      }
      mnew = fmaxf(m_r, rmax);
      float rsum = 0.0f;
      for (int i = 0; i < 32; ++i) {
        float4 t = sr[i];
        rsum += __expf(t.x - mnew) + __expf(t.y - mnew) +
                __expf(t.z - mnew) + __expf(t.w - mnew);
      }
      corr = __expf(m_r - mnew);
      l_r  = l_r * corr + rsum;
      m_r  = mnew;
    }

    // ---- P = exp(S - mnew) as bf16; rescale O by corr ----
#pragma unroll
    for (int v = 0; v < 8; ++v) {
      const int row = v + lh * 8;
      const float mm = __shfl(mnew, row, 32);
      const float cr = __shfl(corr, row, 32);
      Ps[row * 128 + wave * 16 + lr] = f2bf(__expf(c[v] - mm));
#pragma unroll
      for (int j = 0; j < 8; ++j) o[j][v] *= cr;
    }
    __syncthreads();

    // ---- O += P(16x128) . V(128 x d-slice); B-fragments from Vt[b][d][s] ----
#pragma unroll
    for (int j = 0; j < 8; ++j) {
      const size_t d = (size_t)(wave * 128 + j * 16 + lr);
      const unsigned short* vg =
          Vt + ((size_t)b << 22) + (d << 12) + (size_t)(k0 + lh * 16);
#pragma unroll
      for (int kc = 0; kc < 4; ++kc) {
        const unsigned short* pp = Ps + lr * 128 + kc * 32 + lh * 16;
        v16bf a  = frag_ld(pp, pp + 8);
        v16bf bf = frag_ld(vg + kc * 32, vg + kc * 32 + 8);
        o[j] = wmma_bf16(a, bf, o[j]);
      }
    }
  }

  // ---- epilogue: out = O / l (fp32) ----
#pragma unroll
  for (int v = 0; v < 8; ++v) {
    const int row = v + lh * 8;
    const float lsum = __shfl(l_r, row, 32);
    const float inv  = 1.0f / lsum;
#pragma unroll
    for (int j = 0; j < 8; ++j) {
      const int col = wave * 128 + j * 16 + lr;
      out[(((size_t)b * S_LEN + q0 + row) << 10) + col] = o[j][v] * inv;
    }
  }
}

// ---------------------------------------------------------------------------
extern "C" void kernel_launch(void* const* d_in, const int* in_sizes, int n_in,
                              void* d_out, int out_size, void* d_ws, size_t ws_size,
                              hipStream_t stream) {
  const float* inputs = (const float*)d_in[0];
  const int*   length = (const int*)d_in[1];
  const float* Wq     = (const float*)d_in[2];
  const float* Wk     = (const float*)d_in[3];
  const float* Wv     = (const float*)d_in[4];
  // d_in[5] = maxlength (compile-time constant here)
  float* out = (float*)d_out;

  const size_t mat = (size_t)B_DIM * S_LEN * D_DIM;   // 16M bf16 elems = 32 MB
  unsigned short* Qb = (unsigned short*)d_ws;
  unsigned short* Kb = Qb + mat;
  unsigned short* Vt = Kb + mat;                      // stored [b][d][s]

  dim3 g1(D_DIM / 128, (B_DIM * S_LEN) / 128, 3);     // (8, 128, 3)
  qkv_proj_kernel<<<g1, 256, 0, stream>>>(inputs, Wq, Wk, Wv, Qb, Kb, Vt);

  dim3 g2(S_LEN / 16, B_DIM);                          // (256, 4)
  flash_attn_kernel<<<g2, 256, 0, stream>>>(Qb, Kb, Vt, length, out);
}
